// RoPESelfAttention_89455578841412
// MI455X (gfx1250) — matrix operand order repaired and checked by reading, not compiled, not run on hardware
//
#include <hip/hip_runtime.h>
#include <cstdint>

// ---------------------------------------------------------------------------
// RoPE self-attention forward for MI455X (gfx1250, wave32, WMMA).
//   B=2, N=2048, C=1024, H=16, HD=64
// Matrix math: v_wmma_f32_16x16x32_bf16 (f32 accumulate).
// GEMM B-tiles staged to LDS with global_load_async_to_lds_b128 (ASYNCcnt),
// double buffered.  Inputs pre-converted to bf16 once; attention scale folded
// into Q (exact in bf16: power of two), so hot loops are WMMA + loads only.
// ---------------------------------------------------------------------------

typedef __bf16 bf16;
typedef __attribute__((ext_vector_type(16))) __bf16 v16bf;
typedef __attribute__((ext_vector_type(4)))  __bf16 v4bf;
typedef __attribute__((ext_vector_type(8)))  float  v8f;

#define DEVINL __device__ __forceinline__

constexpr int Bb = 2;
constexpr int Nn = 2048;
constexpr int Cc = 1024;
constexpr int Hh = 16;
constexpr int HD = 64;
constexpr int M_ALL = Bb * Nn;          // 4096
constexpr int QKV_N = 3 * Cc;           // 3072
constexpr int LDSP  = 136;              // padded LDS row (bf16), 16B-aligned rows

DEVINL bf16 f2bf(float f) { return (bf16)f; }
DEVINL v8f v8f_zero() {
  v8f z;
#pragma unroll
  for (int i = 0; i < 8; i++) z[i] = 0.f;
  return z;
}
DEVINL v8f wmma_bf16(v16bf a, v16bf b, v8f c) {
  return __builtin_amdgcn_wmma_f32_16x16x32_bf16(false, a, false, b,
                                                 (short)0, c, false, false);
}

// A (16x32, 16-bit): lane m<16: K = {kb..kb+7, 16+kb..16+kb+7}, kb = 8*(lane>=16)
DEVINL v16bf load_a_bf16(const bf16* base, int ld, int lane) {
  int m = lane & 15, kb = (lane >> 4) << 3;
  const bf16* p = base + (size_t)m * ld + kb;
  v16bf a;
#pragma unroll
  for (int i = 0; i < 8; i++) a[i] = p[i];
#pragma unroll
  for (int i = 0; i < 8; i++) a[8 + i] = p[16 + i];
  return a;
}
// B (32x16, 16-bit): lane n<16: column n, K = {kb..kb+7, 16+kb..16+kb+7}, kb = 8*(lane>=16)
DEVINL v16bf load_b_bf16(const bf16* base, int ld, int lane) {
  int n = lane & 15, kb = (lane >> 4) << 3;
  const bf16* p = base + n;
  v16bf b;
#pragma unroll
  for (int i = 0; i < 8; i++) b[i] = p[(size_t)(kb + i) * ld];
#pragma unroll
  for (int i = 0; i < 8; i++) b[8 + i] = p[(size_t)(16 + kb + i) * ld];
  return b;
}

// ===========================================================================
// Kernel 0: f32 -> bf16 one-pass convert (bandwidth bound)
// ===========================================================================
__global__ __launch_bounds__(256)
void cvt_bf16_kernel(const float* __restrict__ in, bf16* __restrict__ out, int n4) {
  int i = blockIdx.x * blockDim.x + threadIdx.x;
  if (i < n4) {
    float4 v = ((const float4*)in)[i];
    v4bf o;
    o[0] = (bf16)v.x; o[1] = (bf16)v.y; o[2] = (bf16)v.z; o[3] = (bf16)v.w;
    *((v4bf*)out + i) = o;
  }
}

// ===========================================================================
// Shared GEMM core: block tile 128(M) x 128(N); 8 waves = 4(M) x 2(N);
// wave tile 32 x 64 (2 A-frags, 8 accumulators, 8 WMMA / K-step).
// B tile (32x128) staged via global_load_async_to_lds_b128, double buffered.
// ===========================================================================
DEVINL void gemm_core(const bf16* __restrict__ A, int lda,
                      const bf16* __restrict__ Bm, int ldb, int K,
                      int row0, int colblk, bf16* sB,
                      v8f acc[2][4], int wn, int lane, int tid) {
  auto stage = [&](int k0, int buf) {
#pragma unroll
    for (int q = 0; q < 2; q++) {
      int idx  = q * 256 + tid;           // 512 x 16B chunks = 8KB
      int rowk = idx >> 4;                // 0..31
      int cc   = (idx & 15) << 3;         // bf16 col, step 8 (16B)
      unsigned lds = (unsigned)(uintptr_t)(sB + ((size_t)(buf * 32 + rowk) * LDSP + cc));
      unsigned long long g =
          (unsigned long long)(uintptr_t)(Bm + (size_t)(k0 + rowk) * ldb + colblk + cc);
      asm volatile("global_load_async_to_lds_b128 %0, %1, off"
                   :: "v"(lds), "v"(g) : "memory");
    }
  };

  stage(0, 0);
  for (int k0 = 0; k0 < K; k0 += 32) {
    const int cur = (k0 >> 5) & 1;
    if (k0 + 32 < K) {
      stage(k0 + 32, cur ^ 1);                           // overlap next tile
      asm volatile("s_wait_asynccnt 0x2" ::: "memory");  // current tile landed
    } else {
      asm volatile("s_wait_asynccnt 0x0" ::: "memory");
    }
    __syncthreads();

    if (k0 + 32 < K)
      __builtin_prefetch(A + (size_t)(row0 + (lane & 15)) * lda + k0 + 32, 0, 1);
    v16bf a0 = load_a_bf16(A + (size_t)row0 * lda + k0, lda, lane);
    v16bf a1 = load_a_bf16(A + (size_t)(row0 + 16) * lda + k0, lda, lane);
    const bf16* bbase = sB + (size_t)cur * 32 * LDSP + wn * 64;
#pragma unroll
    for (int j = 0; j < 4; j++) {
      v16bf b = load_b_bf16(bbase + 16 * j, LDSP, lane);   // ds_load, shared by both M halves
      acc[0][j] = wmma_bf16(a0, b, acc[0][j]);
      acc[1][j] = wmma_bf16(a1, b, acc[1][j]);
    }
    __syncthreads();
  }
}

// ===========================================================================
// Kernel 1: qkv = xb @ wqb + b_qkv, fused RoPE on q/k, scatter to workspace.
//   Q is pre-scaled by HD^-0.5 = 1/8 (exact exponent shift in bf16).
// ===========================================================================
__global__ __launch_bounds__(256)
void qkv_rope_kernel(const bf16* __restrict__ xb,
                     const bf16* __restrict__ wqb,
                     const float* __restrict__ b_qkv,
                     bf16* __restrict__ qb,   // [B,H,N,HD], pre-scaled
                     bf16* __restrict__ kT,   // [B,H,HD,N]
                     bf16* __restrict__ vb)   // [B,H,N,HD]
{
  __shared__ alignas(16) bf16 sB[2 * 32 * LDSP];
  const int tid  = threadIdx.x;
  const int lane = tid & 31, w = tid >> 5;
  const int wm = w & 3, wn = w >> 2;
  const int row0   = blockIdx.y * 128 + wm * 32;
  const int colblk = blockIdx.x * 128;
  const int col0   = colblk + wn * 64;           // 64-aligned => one (sect, head)

  v8f acc[2][4];
#pragma unroll
  for (int mi = 0; mi < 2; mi++)
#pragma unroll
    for (int j = 0; j < 4; j++) acc[mi][j] = v8f_zero();

  gemm_core(xb, Cc, wqb, QKV_N, Cc, row0, colblk, sB, acc, wn, lane, tid);

  const int hi = lane >> 4, nl = lane & 15;
  const int sect = col0 >> 10;                   // 0=q 1=k 2=v
  const int h    = (col0 & 1023) >> 6;
#pragma unroll
  for (int mi = 0; mi < 2; mi++) {
#pragma unroll
    for (int j = 0; j < 4; j++) {
#pragma unroll
      for (int r = 0; r < 8; r++) {
        const int gm = row0 + mi * 16 + r + 8 * hi;
        const int d  = 16 * j + nl;              // d in [0,64)
        const int gc = col0 + d;
        const int bidx = gm >> 11, n = gm & 2047;
        float val = acc[mi][j][r] + b_qkv[gc];
        if (sect == 2) {
          vb[(((size_t)(bidx * Hh + h) * Nn) + n) * HD + d] = f2bf(val);
        } else {
          float pv  = acc[mi][j ^ 2][r] + b_qkv[gc ^ 32];
          float rot = (d < 32) ? -pv : pv;
          float ang = (float)n * __expf(-0.2878231366242335f * (float)(d & 31));
          float cs, sn;
          __sincosf(ang, &sn, &cs);
          float outv = val * cs + rot * sn;
          if (sect == 0)
            qb[(((size_t)(bidx * Hh + h) * Nn) + n) * HD + d] = f2bf(outv * 0.125f);
          else
            kT[(((size_t)(bidx * Hh + h) * HD) + d) * Nn + n] = f2bf(outv);
        }
      }
    }
  }
}

// ===========================================================================
// Kernel 2: flash attention. One wave = 16 q rows; block = 128 rows.
//   S and O in accumulators; P converted C->A layout via wave-private LDS.
//   Q already carries the 1/8 scale.
// ===========================================================================
__global__ __launch_bounds__(256)
void flash_attn_kernel(const bf16* __restrict__ qb,
                       const bf16* __restrict__ kT,
                       const bf16* __restrict__ vb,
                       bf16* __restrict__ ob)   // [B*N, C]
{
  __shared__ alignas(16) bf16 plds[8][16][64];

  const int lane = threadIdx.x & 31;
  const int w    = threadIdx.x >> 5;
  const int bh   = blockIdx.y;
  const int b    = bh >> 4, h = bh & 15;
  const int qrow0 = blockIdx.x * 128 + w * 16;

  const bf16* qh = qb + (size_t)bh * Nn * HD;
  const bf16* kh = kT + (size_t)bh * HD * Nn;
  const bf16* vh = vb + (size_t)bh * Nn * HD;

  const v16bf aq0 = load_a_bf16(qh + (size_t)qrow0 * HD, HD, lane);
  const v16bf aq1 = load_a_bf16(qh + (size_t)qrow0 * HD + 32, HD, lane);

  v8f o[4];
#pragma unroll
  for (int j = 0; j < 4; j++) o[j] = v8f_zero();
  float mrow[8], lrow[8];
#pragma unroll
  for (int r = 0; r < 8; r++) { mrow[r] = -1e30f; lrow[r] = 0.f; }

  const int hi = lane >> 4, nl = lane & 15;

  for (int kb0 = 0; kb0 < Nn; kb0 += 64) {
    v8f s[4];
#pragma unroll
    for (int j = 0; j < 4; j++) {
      v16bf bk0 = load_b_bf16(kh + kb0 + 16 * j, Nn, lane);                   // d 0..31
      s[j] = wmma_bf16(aq0, bk0, v8f_zero());
      v16bf bk1 = load_b_bf16(kh + (size_t)32 * Nn + kb0 + 16 * j, Nn, lane); // d 32..63
      s[j] = wmma_bf16(aq1, bk1, s[j]);
    }

#pragma unroll
    for (int r = 0; r < 8; r++) {
      float v = fmaxf(fmaxf(s[0][r], s[1][r]), fmaxf(s[2][r], s[3][r]));
#pragma unroll
      for (int msk = 1; msk < 16; msk <<= 1) v = fmaxf(v, __shfl_xor(v, msk, 32));
      float mn = fmaxf(mrow[r], v);
      float corr = __expf(mrow[r] - mn);
      mrow[r] = mn;
      lrow[r] *= corr;
#pragma unroll
      for (int j = 0; j < 4; j++) o[j][r] *= corr;
      float rsum = 0.f;
#pragma unroll
      for (int j = 0; j < 4; j++) {
        float p = __expf(s[j][r] - mn);
        s[j][r] = p;
        rsum += p;
      }
#pragma unroll
      for (int msk = 1; msk < 16; msk <<= 1) rsum += __shfl_xor(rsum, msk, 32);
      lrow[r] += rsum;
    }

    // P: C-layout regs -> LDS (16x64 row-major) -> A-layout frags (wave-private)
#pragma unroll
    for (int j = 0; j < 4; j++)
#pragma unroll
      for (int r = 0; r < 8; r++)
        plds[w][r + 8 * hi][16 * j + nl] = f2bf(s[j][r]);

    v16bf pa0 = load_a_bf16(&plds[w][0][0], 64, lane);
    v16bf pa1 = load_a_bf16(&plds[w][0][0] + 32, 64, lane);

#pragma unroll
    for (int j = 0; j < 4; j++) {
      v16bf bv0 = load_b_bf16(vh + (size_t)kb0 * HD + 16 * j, HD, lane);
      o[j] = wmma_bf16(pa0, bv0, o[j]);
      v16bf bv1 = load_b_bf16(vh + (size_t)(kb0 + 32) * HD + 16 * j, HD, lane);
      o[j] = wmma_bf16(pa1, bv1, o[j]);
    }
  }

#pragma unroll
  for (int r = 0; r < 8; r++) {
    float inv = 1.f / lrow[r];
    const int gm = b * Nn + qrow0 + r + 8 * hi;
#pragma unroll
    for (int j = 0; j < 4; j++) {
      const int gc = h * HD + 16 * j + nl;
      ob[(size_t)gm * Cc + gc] = f2bf(o[j][r] * inv);
    }
  }
}

// ===========================================================================
// Kernel 3: out = ob @ wpb + b_proj   (f32 out)
// ===========================================================================
__global__ __launch_bounds__(256)
void proj_kernel(const bf16* __restrict__ ob,
                 const bf16* __restrict__ wpb,
                 const float* __restrict__ b_proj,
                 float* __restrict__ out)
{
  __shared__ alignas(16) bf16 sB[2 * 32 * LDSP];
  const int tid  = threadIdx.x;
  const int lane = tid & 31, w = tid >> 5;
  const int wm = w & 3, wn = w >> 2;
  const int row0   = blockIdx.y * 128 + wm * 32;
  const int colblk = blockIdx.x * 128;
  const int col0   = colblk + wn * 64;

  v8f acc[2][4];
#pragma unroll
  for (int mi = 0; mi < 2; mi++)
#pragma unroll
    for (int j = 0; j < 4; j++) acc[mi][j] = v8f_zero();

  gemm_core(ob, Cc, wpb, Cc, Cc, row0, colblk, sB, acc, wn, lane, tid);

  const int hi = lane >> 4, nl = lane & 15;
#pragma unroll
  for (int mi = 0; mi < 2; mi++)
#pragma unroll
    for (int j = 0; j < 4; j++)
#pragma unroll
      for (int r = 0; r < 8; r++) {
        const int gm = row0 + mi * 16 + r + 8 * hi;
        const int gc = col0 + 16 * j + nl;
        out[(size_t)gm * Cc + gc] = acc[mi][j][r] + b_proj[gc];
      }
}

// ===========================================================================
extern "C" void kernel_launch(void* const* d_in, const int* in_sizes, int n_in,
                              void* d_out, int out_size, void* d_ws, size_t ws_size,
                              hipStream_t stream) {
  const float* x      = (const float*)d_in[0];
  const float* w_qkv  = (const float*)d_in[1];
  const float* b_qkv  = (const float*)d_in[2];
  const float* w_proj = (const float*)d_in[3];
  const float* b_proj = (const float*)d_in[4];
  float* out = (float*)d_out;

  const size_t per = (size_t)Bb * Hh * Nn * HD;   // 4,194,304
  bf16* qb  = (bf16*)d_ws;
  bf16* kT  = qb + per;
  bf16* vb  = kT + per;
  bf16* ob  = vb + per;                           // [B*N, C]
  bf16* xb  = ob + per;                           // [B*N, C]
  bf16* wqb = xb + per;                           // [C, 3C]
  bf16* wpb = wqb + (size_t)Cc * QKV_N;           // [C, C]

  cvt_bf16_kernel<<<(M_ALL * Cc / 4) / 256, 256, 0, stream>>>(x, xb, M_ALL * Cc / 4);
  cvt_bf16_kernel<<<(Cc * QKV_N / 4) / 256, 256, 0, stream>>>(w_qkv, wqb, Cc * QKV_N / 4);
  cvt_bf16_kernel<<<(Cc * Cc / 4) / 256, 256, 0, stream>>>(w_proj, wpb, Cc * Cc / 4);

  {  // qkv + rope (pre-scales q by 1/8)
    dim3 grid(QKV_N / 128, M_ALL / 128);          // (24, 32)
    qkv_rope_kernel<<<grid, 256, 0, stream>>>(xb, wqb, b_qkv, qb, kT, vb);
  }
  {  // flash attention
    dim3 grid(Nn / 128, Bb * Hh);                 // (16, 32)
    flash_attn_kernel<<<grid, 256, 0, stream>>>(qb, kT, vb, ob);
  }
  {  // output projection
    dim3 grid(Cc / 128, M_ALL / 128);             // (8, 32)
    proj_kernel<<<grid, 256, 0, stream>>>(ob, wpb, b_proj, out);
  }
}
